// MultiHeadAttention_61418032333230
// MI455X (gfx1250) — matrix operand order repaired and checked by reading, not compiled, not run on hardware
//
#include <hip/hip_runtime.h>

// ---------------------------------------------------------------------------
// MI455X (gfx1250) multi-head attention, bf16 WMMA path.
//   SEQ=4096, DIM=1024, 16 heads x head_dim 64, causal mask applied AFTER
//   softmax (denominator includes future keys), matching the reference.
// ---------------------------------------------------------------------------

typedef __attribute__((ext_vector_type(16))) __bf16 bf16x16;
typedef __attribute__((ext_vector_type(8)))  __bf16 bf16x8;
typedef __attribute__((ext_vector_type(4)))  __bf16 bf16x4;
typedef __attribute__((ext_vector_type(8)))  float  f32x8;

#define SEQ    4096
#define DIM    1024
#define NHEADS 16
#define HD     64

// D = A(16x32 bf16) x B(32x16 bf16) + C(16x16 f32)
static __device__ __forceinline__ f32x8 wmma_bf16(bf16x16 a, bf16x16 b, f32x8 c) {
  return __builtin_amdgcn_wmma_f32_16x16x32_bf16(false, a, false, b, (short)0, c,
                                                 false, false);
}

// A-operand (16x32, 16-bit) from row-major data. Per ISA 7.12.2:
//   lanes 0-15 : elems 0-7 => K = k0+0..7,   elems 8-15 => K = k0+16..23
//   lanes16-31 : elems 0-7 => K = k0+8..15,  elems 8-15 => K = k0+24..31
static __device__ __forceinline__ bf16x16 load_afrag(const __bf16* row, int k0, int half) {
  bf16x8 lo = *(const bf16x8*)(row + k0 + 8 * half);
  bf16x8 hi = *(const bf16x8*)(row + k0 + 16 + 8 * half);
  bf16x16 r;
#pragma unroll
  for (int i = 0; i < 8; ++i) { r[i] = lo[i]; r[8 + i] = hi[i]; }
  return r;
}

// B-operand (32x16, 16-bit): lane holds one column; same K order as A:
// lanes 0-15 => K=k0+0..7,16..23; lanes 16-31 => K=k0+8..15,24..31.
static __device__ __forceinline__ bf16x16 load_bfrag(const __bf16* row, int k0, int half) {
  return load_afrag(row, k0, half);
}

// lane L <-> lane L^16 exchange (SWAPX16 group-of-32 swizzle)
static __device__ __forceinline__ float xor16(float v) {
  return __builtin_bit_cast(float,
      __builtin_amdgcn_ds_swizzle(__builtin_bit_cast(int, v), 0x401F));
}
// broadcast value of lane `srclane` (wave32)
static __device__ __forceinline__ float bcast(float v, int srclane) {
  return __builtin_bit_cast(float,
      __builtin_amdgcn_ds_bpermute(srclane << 2, __builtin_bit_cast(int, v)));
}

// ---------------------------------------------------------------------------
// Phase 0a: fp32 content -> bf16 copy
// ---------------------------------------------------------------------------
__global__ __launch_bounds__(256) void convert_x_kernel(const float* __restrict__ x,
                                                        __bf16* __restrict__ xb) {
  const size_t i = ((size_t)blockIdx.x * blockDim.x + threadIdx.x) * 4;
  float4 v = *(const float4*)(x + i);
  bf16x4 o = { (__bf16)v.x, (__bf16)v.y, (__bf16)v.z, (__bf16)v.w };
  *(bf16x4*)(xb + i) = o;
}

// ---------------------------------------------------------------------------
// Phase 0b: W (KxN fp32 row-major) -> Wt (NxK bf16 row-major) via LDS tiles
// ---------------------------------------------------------------------------
__global__ __launch_bounds__(256) void transpose_w_kernel(const float* __restrict__ w,
                                                          __bf16* __restrict__ wt) {
  __shared__ float tile[32][33];
  const int n0 = blockIdx.x * 32;
  const int k0 = blockIdx.y * 32;
  for (int j = threadIdx.y; j < 32; j += 8)
    tile[j][threadIdx.x] = w[(size_t)(k0 + j) * DIM + n0 + threadIdx.x];
  __syncthreads();
  for (int j = threadIdx.y; j < 32; j += 8)
    wt[(size_t)(n0 + j) * DIM + k0 + threadIdx.x] = (__bf16)tile[threadIdx.x][j];
}

// ---------------------------------------------------------------------------
// Phase 1: fused QKV projection.
// One wave per (16-row M tile, 64-col head strip): 12 f32x8 accumulators
// (Q,K,V x 4 sub-tiles) share every A fragment. Q stored pre-scaled by
// log2(e)/sqrt(64); V stored transposed as Vt[head][dim][seq].
// ---------------------------------------------------------------------------
__global__ __launch_bounds__(128) void qkv_proj_kernel(
    const __bf16* __restrict__ Xb,
    const __bf16* __restrict__ Wtq, const __bf16* __restrict__ Wtk,
    const __bf16* __restrict__ Wtv,
    __bf16* __restrict__ Qb, __bf16* __restrict__ Kbuf, __bf16* __restrict__ Vt) {
  const int lane = threadIdx.x & 31;
  const int wid  = threadIdx.x >> 5;
  const int l15  = lane & 15;
  const int half = lane >> 4;
  const int task  = blockIdx.x * 4 + wid;   // 4096 wave-tasks
  const int mtile = task >> 4;              // 0..255
  const int h     = task & 15;              // head == 64-wide N strip

  const __bf16* xrow = Xb + (size_t)(mtile * 16 + l15) * DIM;
  const __bf16* wbase[3] = { Wtq, Wtk, Wtv };
  const __bf16* wrow[3][4];
#pragma unroll
  for (int s = 0; s < 3; ++s)
#pragma unroll
    for (int sub = 0; sub < 4; ++sub)
      wrow[s][sub] = wbase[s] + (size_t)(h * 64 + sub * 16 + l15) * DIM;

  f32x8 acc[3][4];
#pragma unroll
  for (int s = 0; s < 3; ++s)
#pragma unroll
    for (int sub = 0; sub < 4; ++sub) acc[s][sub] = (f32x8){};

  for (int k0 = 0; k0 < DIM; k0 += 32) {
    bf16x16 a = load_afrag(xrow, k0, half);
#pragma unroll
    for (int s = 0; s < 3; ++s)
#pragma unroll
      for (int sub = 0; sub < 4; ++sub)
        acc[s][sub] = wmma_bf16(a, load_bfrag(wrow[s][sub], k0, half), acc[s][sub]);
  }

  const float qscale = 0.18033688011112042f;  // log2(e) / sqrt(HEAD_DIM)
  const int m0 = mtile * 16 + half * 8;
#pragma unroll
  for (int sub = 0; sub < 4; ++sub) {
    const int d = sub * 16 + l15;
#pragma unroll
    for (int r = 0; r < 8; ++r) {
      const int mrow = m0 + r;
      Qb  [(size_t)(h * SEQ + mrow) * HD + d] = (__bf16)(acc[0][sub][r] * qscale);
      Kbuf[(size_t)(h * SEQ + mrow) * HD + d] = (__bf16)(acc[1][sub][r]);
      Vt  [(size_t)(h * HD + d) * SEQ + mrow] = (__bf16)(acc[2][sub][r]);
    }
  }
}

// ---------------------------------------------------------------------------
// Phase 2: flash attention (full-row softmax denominator, causal PV).
// One wave handles 16 queries of one head. Scores computed TRANSPOSED
// (S^T = K_blk x Q^T) so the C-layout feeds the P.V A-operand lane-locally.
// ---------------------------------------------------------------------------
__global__ __launch_bounds__(256) void attn_kernel(
    const __bf16* __restrict__ Qb, const __bf16* __restrict__ Kb,
    const __bf16* __restrict__ Vt, float* __restrict__ out) {
  const int lane = threadIdx.x & 31;
  const int wid  = threadIdx.x >> 5;
  const int l15  = lane & 15;
  const int half = lane >> 4;
  const int task = blockIdx.x * 8 + wid;    // 4096 wave-tasks
  const int h    = task >> 8;
  const int qm0  = (task & 255) << 4;

  const __bf16* qrow = Qb + (size_t)(h * SEQ + qm0 + l15) * HD;
  const bf16x16 qf0 = load_bfrag(qrow, 0, half);
  const bf16x16 qf1 = load_bfrag(qrow, 32, half);

  const __bf16* kbase = Kb + (size_t)h * SEQ * HD;
  const __bf16* vbase = Vt + (size_t)h * HD * SEQ;
  const __bf16* vrow0 = vbase + (size_t)(l15) * SEQ;
  const __bf16* vrow1 = vbase + (size_t)(16 + l15) * SEQ;
  const __bf16* vrow2 = vbase + (size_t)(32 + l15) * SEQ;
  const __bf16* vrow3 = vbase + (size_t)(48 + l15) * SEQ;

  f32x8 o0 = {}, o1 = {}, o2 = {}, o3 = {};
  float m = -__builtin_inff(), l = 0.f, mL = 0.f;

  for (int kb = 0; kb < SEQ; kb += 32) {
    const __bf16* kr0 = kbase + (size_t)(kb + l15) * HD;
    const __bf16* kr1 = kr0 + 16 * HD;
    f32x8 st0 = {}, st1 = {};
    st0 = wmma_bf16(load_afrag(kr0, 0, half),  qf0, st0);
    st0 = wmma_bf16(load_afrag(kr0, 32, half), qf1, st0);
    st1 = wmma_bf16(load_afrag(kr1, 0, half),  qf0, st1);
    st1 = wmma_bf16(load_afrag(kr1, 32, half), qf1, st1);

    // per-query (lane l15) running max / sum over all 32 keys of the block
    float mb = st0[0];
#pragma unroll
    for (int r = 0; r < 8; ++r) { mb = fmaxf(mb, st0[r]); mb = fmaxf(mb, st1[r]); }
    mb = fmaxf(mb, xor16(mb));
    const float mn = fmaxf(m, mb);
    const float sc = __builtin_amdgcn_exp2f(m - mn);
    float p0[8], p1[8];
    float lb = 0.f;
#pragma unroll
    for (int r = 0; r < 8; ++r) {
      p0[r] = __builtin_amdgcn_exp2f(st0[r] - mn);
      p1[r] = __builtin_amdgcn_exp2f(st1[r] - mn);
      lb += p0[r] + p1[r];
    }
    lb += xor16(lb);
    l = l * sc + lb;   // denominator counts masked (future) keys, per reference
    m = mn;

    if (kb <= qm0 + 15) {                    // block contributes to P.V
      if (kb + 31 > qm0) {                   // diagonal block: mask P only
        const int q = qm0 + l15;
#pragma unroll
        for (int r = 0; r < 8; ++r) {
          if (kb + 8 * half + r      > q) p0[r] = 0.f;
          if (kb + 16 + 8 * half + r > q) p1[r] = 0.f;
        }
      }
      const int hb = half << 3;
#pragma unroll
      for (int r = 0; r < 8; ++r) {          // O row-scale via lane broadcast
        const float f = bcast(sc, hb + r);
        o0[r] *= f; o1[r] *= f; o2[r] *= f; o3[r] *= f;
      }
      bf16x16 pa;                            // S^T C-layout -> A-operand, lane-local
#pragma unroll
      for (int r = 0; r < 8; ++r) { pa[r] = (__bf16)p0[r]; pa[8 + r] = (__bf16)p1[r]; }
      o0 = wmma_bf16(pa, load_bfrag(vrow0, kb, half), o0);
      o1 = wmma_bf16(pa, load_bfrag(vrow1, kb, half), o1);
      o2 = wmma_bf16(pa, load_bfrag(vrow2, kb, half), o2);
      o3 = wmma_bf16(pa, load_bfrag(vrow3, kb, half), o3);
      mL = mn;
    }
  }

  const float fl = __builtin_amdgcn_exp2f(mL - m) / l;  // deferred scale + 1/denom
  const int hb = half << 3;
#pragma unroll
  for (int r = 0; r < 8; ++r) {
    const float fr = bcast(fl, hb + r);
    float* orow = out + (size_t)(qm0 + hb + r) * DIM + h * HD;
    orow[l15]      = o0[r] * fr;
    orow[16 + l15] = o1[r] * fr;
    orow[32 + l15] = o2[r] * fr;
    orow[48 + l15] = o3[r] * fr;
  }
}

// ---------------------------------------------------------------------------
extern "C" void kernel_launch(void* const* d_in, const int* in_sizes, int n_in,
                              void* d_out, int out_size, void* d_ws, size_t ws_size,
                              hipStream_t stream) {
  const float* X  = (const float*)d_in[0];
  const float* Wq = (const float*)d_in[1];
  const float* Wk = (const float*)d_in[2];
  const float* Wv = (const float*)d_in[3];
  float* out = (float*)d_out;

  char* w = (char*)d_ws;                       // 38 MB used
  __bf16* Xb  = (__bf16*)(w);                  // 8 MB  [4096][1024]
  __bf16* Wtq = (__bf16*)(w + (8u  << 20));    // 2 MB  [1024][1024] (transposed)
  __bf16* Wtk = (__bf16*)(w + (10u << 20));    // 2 MB
  __bf16* Wtv = (__bf16*)(w + (12u << 20));    // 2 MB
  __bf16* Qb  = (__bf16*)(w + (14u << 20));    // 8 MB  [head][seq][64], pre-scaled
  __bf16* Kb  = (__bf16*)(w + (22u << 20));    // 8 MB  [head][seq][64]
  __bf16* Vt  = (__bf16*)(w + (30u << 20));    // 8 MB  [head][64][seq]

  convert_x_kernel<<<4096, 256, 0, stream>>>(X, Xb);
  dim3 tb(32, 8), tg(32, 32);
  transpose_w_kernel<<<tg, tb, 0, stream>>>(Wq, Wtq);
  transpose_w_kernel<<<tg, tb, 0, stream>>>(Wk, Wtk);
  transpose_w_kernel<<<tg, tb, 0, stream>>>(Wv, Wtv);
  qkv_proj_kernel<<<1024, 128, 0, stream>>>(Xb, Wtq, Wtk, Wtv, Qb, Kb, Vt);
  attn_kernel<<<512, 256, 0, stream>>>(Qb, Kb, Vt, out);
}
